// MayaVivekaNet_28260884807875
// MI455X (gfx1250) — matrix operand order repaired and checked by reading, not compiled, not run on hardware
//
#include <hip/hip_runtime.h>
#include <hip/hip_bf16.h>

// ---------------------------------------------------------------------------
// MI455X (gfx1250) spiking CNN forward.
// Spikes are exactly {0,1} -> f16 WMMA (v_wmma_f32_16x16x32_f16, f32 accum)
// is lossless on activations. BN folded into f16 weights + f32 bias.
// conv2/conv3: shifted implicit GEMM, LDS halo staging, wave32 WMMA tiles.
// ---------------------------------------------------------------------------

typedef _Float16 half_t;
typedef __attribute__((ext_vector_type(8)))  _Float16 v8h;
typedef __attribute__((ext_vector_type(16))) _Float16 v16h;
typedef __attribute__((ext_vector_type(8)))  float    v8f;

__device__ __forceinline__ v16h cat16(v8h lo, v8h hi) {
  return __builtin_shufflevector(lo, hi, 0,1,2,3,4,5,6,7,8,9,10,11,12,13,14,15);
}

// ---------------------------------------------------------------------------
// Prep: fold BN scale into conv weights -> f16 [tap][COUT][CIN]; bias f32.
// ---------------------------------------------------------------------------
template<int CIN, int COUT>
__global__ void prep_conv(const float* __restrict__ w,
                          const float* __restrict__ g, const float* __restrict__ b,
                          const float* __restrict__ m, const float* __restrict__ var,
                          half_t* __restrict__ wf, float* __restrict__ bias) {
  int idx = blockIdx.x * blockDim.x + threadIdx.x;
  const int total = 9 * COUT * CIN;
  if (idx < total) {
    int ci  = idx % CIN;
    int co  = (idx / CIN) % COUT;
    int tap = idx / (CIN * COUT);
    float s = g[co] * rsqrtf(var[co] + 1e-5f);
    wf[idx] = (half_t)(w[(co * CIN + ci) * 9 + tap] * s);
  }
  if (idx < COUT) {
    float s = g[idx] * rsqrtf(var[idx] + 1e-5f);
    bias[idx] = b[idx] - m[idx] * s;
  }
}

// Permute FC weights: d = c*64 + (h*8+w)  ->  d' = (h*8+w)*256 + c, to f16.
__global__ void prep_fc(const float* __restrict__ fcw, half_t* __restrict__ out) {
  int idx = blockIdx.x * blockDim.x + threadIdx.x;   // < 512*16384
  int f  = idx >> 14;
  int dp = idx & 16383;
  int hw = dp >> 8;     // 0..63
  int c  = dp & 255;
  out[idx] = (half_t)fcw[(f << 14) + c * 64 + hw];
}

// ---------------------------------------------------------------------------
// conv1: 3->64, K=27 is tiny -> direct VALU conv, weights staged in LDS.
// Input x: [TB,3,32,32] f32 (NCHW). Output preact: [TB,32,32,64] f16.
// One thread = one (n,y,x,co8) -> 8 channels, single b128 store.
// ---------------------------------------------------------------------------
__global__ __launch_bounds__(256)
void conv1_direct(const float* __restrict__ x, const half_t* __restrict__ wf,
                  const float* __restrict__ bias, half_t* __restrict__ preact) {
  __shared__ float wsh[9 * 64 * 3];
  __shared__ float bsh[64];
  for (int i = threadIdx.x; i < 9 * 64 * 3; i += blockDim.x) wsh[i] = (float)wf[i];
  if (threadIdx.x < 64) bsh[threadIdx.x] = bias[threadIdx.x];
  __syncthreads();

  long g  = (long)blockIdx.x * blockDim.x + threadIdx.x;
  int co8 = (int)(g & 7);
  int xx  = (int)((g >> 3) & 31);
  int yy  = (int)((g >> 8) & 31);
  int n   = (int)(g >> 13);
  const float* xn = x + (long)n * 3 * 1024;

  float acc[8];
#pragma unroll
  for (int r = 0; r < 8; r++) acc[r] = bsh[co8 * 8 + r];

#pragma unroll
  for (int ky = 0; ky < 3; ky++) {
    int y2 = yy + ky - 1;
    if ((unsigned)y2 >= 32u) continue;
#pragma unroll
    for (int kx = 0; kx < 3; kx++) {
      int x2 = xx + kx - 1;
      if ((unsigned)x2 >= 32u) continue;
      int tap = ky * 3 + kx;
#pragma unroll
      for (int ci = 0; ci < 3; ci++) {
        float xv = xn[ci * 1024 + y2 * 32 + x2];
#pragma unroll
        for (int r = 0; r < 8; r++)
          acc[r] += xv * wsh[(tap * 64 + co8 * 8 + r) * 3 + ci];
      }
    }
  }
  v8h o;
#pragma unroll
  for (int r = 0; r < 8; r++) o[r] = (half_t)acc[r];
  *reinterpret_cast<v8h*>(preact + g * 8) = o;
}

// ---------------------------------------------------------------------------
// LIF scan, no pool: per (b,h,w,c8) element, loop t. v = 0.5v+0.5x; hard reset.
// ---------------------------------------------------------------------------
__global__ void lif_nopool(const half_t* __restrict__ pre, half_t* __restrict__ spk,
                           long stride_t) {
  long i = ((long)blockIdx.x * blockDim.x + threadIdx.x) * 8;
  float v[8];
#pragma unroll
  for (int r = 0; r < 8; r++) v[r] = 0.f;
  for (int t = 0; t < 8; t++) {
    v8h xv = *reinterpret_cast<const v8h*>(pre + i + (long)t * stride_t);
    v8h s;
#pragma unroll
    for (int r = 0; r < 8; r++) {
      float vv = v[r] + ((float)xv[r] - v[r]) * 0.5f;
      float sp = (vv >= 1.f) ? 1.f : 0.f;
      v[r] = vv * (1.f - sp);
      s[r] = (half_t)sp;
    }
    *reinterpret_cast<v8h*>(spk + i + (long)t * stride_t) = s;
  }
}

// ---------------------------------------------------------------------------
// LIF + 2x2 maxpool fused (binary max == OR). One thread per (b,yo,xo,c8),
// keeps 4 membrane states, loops t, stores pooled spikes per t.
// ---------------------------------------------------------------------------
template<int C, int HIN>
__global__ void lif_pool(const half_t* __restrict__ pre, half_t* __restrict__ spk) {
  constexpr int HO = HIN / 2;
  constexpr int C8 = C / 8;
  long g  = (long)blockIdx.x * blockDim.x + threadIdx.x;
  int c8  = (int)(g % C8);
  long r1 = g / C8;
  int xo  = (int)(r1 % HO);
  long r2 = r1 / HO;
  int yo  = (int)(r2 % HO);
  int b   = (int)(r2 / HO);
  const long HWCin  = (long)HIN * HIN * C;
  const long HWCout = (long)HO * HO * C;

  float v[4][8];
#pragma unroll
  for (int q = 0; q < 4; q++)
#pragma unroll
    for (int r = 0; r < 8; r++) v[q][r] = 0.f;

  for (int t = 0; t < 8; t++) {
    long nin = (long)(t * 128 + b);
    float smax[8];
#pragma unroll
    for (int r = 0; r < 8; r++) smax[r] = 0.f;
#pragma unroll
    for (int dy = 0; dy < 2; dy++) {
#pragma unroll
      for (int dx = 0; dx < 2; dx++) {
        int q = dy * 2 + dx;
        v8h xv = *reinterpret_cast<const v8h*>(
            pre + nin * HWCin + (((long)(2 * yo + dy) * HIN) + (2 * xo + dx)) * C + c8 * 8);
#pragma unroll
        for (int r = 0; r < 8; r++) {
          float vv = v[q][r] + ((float)xv[r] - v[q][r]) * 0.5f;
          float sp = (vv >= 1.f) ? 1.f : 0.f;
          v[q][r] = vv * (1.f - sp);
          smax[r] = fmaxf(smax[r], sp);
        }
      }
    }
    v8h o;
#pragma unroll
    for (int r = 0; r < 8; r++) o[r] = (half_t)smax[r];
    *reinterpret_cast<v8h*>(spk + nin * HWCout + ((long)yo * HO + xo) * C + c8 * 8) = o;
  }
}

// ---------------------------------------------------------------------------
// 3x3 SAME conv as shifted implicit GEMM with WMMA f16 (f32 accum).
// in : [NT,H,W,CIN] f16 spikes, wf: [9][COUT][CIN] f16 (BN folded),
// out: [NT,H,W,COUT] f16 preact (+bias).
// Block = 256 thr = 8 waves. Block tile: 16 pixels (one row seg) x 128 Cout.
// LDS: 3x18xCIN halo tile (zero-padded). Per wave: 9 taps * CIN/32 WMMAs.
// A frag: lanes 0-15 M=co,K{0..7,16..23}; lanes 16-31 K{8..15,24..31}.
// B frag: lane L -> N=L&15, same K order as A frag.
// ---------------------------------------------------------------------------
template<int CIN, int COUT, int H, int W>
__global__ __launch_bounds__(256)
void conv_wmma(const half_t* __restrict__ in, const half_t* __restrict__ wf,
               const float* __restrict__ bias, half_t* __restrict__ out) {
  constexpr int XT = W / 16;
  __shared__ __attribute__((aligned(16))) half_t tile[3][18][CIN];

  int bid = blockIdx.x;
  int xt = bid % XT;
  int y  = (bid / XT) % H;
  int n  = bid / (XT * H);
  int x0 = xt * 16;
  int tid = threadIdx.x;

  // Stage zero-padded halo tile into LDS (b128 per item).
  const int ITEMS = 3 * 18 * (CIN / 8);
  for (int it = tid; it < ITEMS; it += 256) {
    int c8 = it % (CIN / 8);
    int lx = (it / (CIN / 8)) % 18;
    int dy = it / (18 * (CIN / 8));
    int yy = y + dy - 1;
    int xx = x0 + lx - 1;
    v8h val;
    if ((unsigned)yy < (unsigned)H && (unsigned)xx < (unsigned)W) {
      val = *reinterpret_cast<const v8h*>(in + (((long)n * H + yy) * W + xx) * CIN + c8 * 8);
    } else {
#pragma unroll
      for (int r = 0; r < 8; r++) val[r] = (half_t)0.f;
    }
    *reinterpret_cast<v8h*>(&tile[dy][lx][c8 * 8]) = val;
  }
  __syncthreads();

  int lane = tid & 31;
  int wave = tid >> 5;
  int co0  = blockIdx.y * 128 + wave * 16;
  int hi   = (lane < 16) ? 0 : 1;
  int px   = lane & 15;

  v8f acc = {};
#pragma unroll
  for (int dy = 0; dy < 3; dy++) {
#pragma unroll
    for (int dx = 0; dx < 3; dx++) {
      const half_t* wbase = wf + (((long)(dy * 3 + dx) * COUT) + co0 + px) * CIN;
      const half_t* bcol  = &tile[dy][px + dx][0];
#pragma unroll
      for (int kc = 0; kc < CIN / 32; kc++) {
        v8h alo = *reinterpret_cast<const v8h*>(wbase + kc * 32 + hi * 8);
        v8h ahi = *reinterpret_cast<const v8h*>(wbase + kc * 32 + 16 + hi * 8);
        v8h blo = *reinterpret_cast<const v8h*>(bcol + kc * 32 + hi * 8);
        v8h bhi = *reinterpret_cast<const v8h*>(bcol + kc * 32 + 16 + hi * 8);
        acc = __builtin_amdgcn_wmma_f32_16x16x32_f16(false, cat16(alo, ahi),
                                                     false, cat16(blo, bhi),
                                                     (short)0, acc, false, false);
      }
    }
  }

  // + bias, store 8 contiguous channels as one b128.
  int xp = x0 + px;
  const float* bb = bias + co0 + hi * 8;
  v8h o;
#pragma unroll
  for (int r = 0; r < 8; r++) o[r] = (half_t)(acc[r] + bb[r]);
  *reinterpret_cast<v8h*>(out + (((long)n * H + y) * W + xp) * COUT + co0 + hi * 8) = o;
}

// ---------------------------------------------------------------------------
// FC: [1024,16384] x [512,16384]^T -> [1024,512] f32, WMMA, K loop of 512.
// ---------------------------------------------------------------------------
__global__ __launch_bounds__(256)
void fc_wmma(const half_t* __restrict__ a, const half_t* __restrict__ w,
             float* __restrict__ out) {
  int lane = threadIdx.x & 31;
  int wave = threadIdx.x >> 5;
  int n0 = blockIdx.x * 16;
  int f0 = blockIdx.y * 128 + wave * 16;
  int hi  = (lane < 16) ? 0 : 1;
  int l15 = lane & 15;
  const half_t* abase = a + (long)(n0 + l15) * 16384;
  const half_t* wbase = w + (long)(f0 + l15) * 16384;
  v8f acc = {};
  for (int d0 = 0; d0 < 16384; d0 += 32) {
    __builtin_prefetch(abase + d0 + 1024, 0, 1);   // global_prefetch hint
    v8h alo = *reinterpret_cast<const v8h*>(abase + d0 + hi * 8);
    v8h ahi = *reinterpret_cast<const v8h*>(abase + d0 + 16 + hi * 8);
    v8h blo = *reinterpret_cast<const v8h*>(wbase + d0 + hi * 8);
    v8h bhi = *reinterpret_cast<const v8h*>(wbase + d0 + 16 + hi * 8);
    acc = __builtin_amdgcn_wmma_f32_16x16x32_f16(false, cat16(alo, ahi),
                                                 false, cat16(blo, bhi),
                                                 (short)0, acc, false, false);
  }
#pragma unroll
  for (int r = 0; r < 8; r++)
    out[(long)(n0 + r + hi * 8) * 512 + f0 + l15] = acc[r];
}

// LIF over FC output + mean over T. pre: [(t*128+b),512] f32 -> feat [128,512].
__global__ void lif_fc_mean(const float* __restrict__ pre, float* __restrict__ feat) {
  int i = blockIdx.x * blockDim.x + threadIdx.x;  // b*512 + f
  float v = 0.f, acc = 0.f;
  for (int t = 0; t < 8; t++) {
    float x = pre[(long)t * 65536 + i];
    v = v + (x - v) * 0.5f;
    float s = (v >= 1.f) ? 1.f : 0.f;
    acc += s;
    v *= (1.f - s);
  }
  feat[i] = acc * 0.125f;
}

// Cosine logits vs prototypes: one wave32 per batch row.
__global__ void logits_kernel(const float* __restrict__ feat,
                              const float* __restrict__ protos,
                              float* __restrict__ out) {
  int b = blockIdx.x;
  int lane = threadIdx.x;
  float fs = 0.f;
  for (int f = lane; f < 512; f += 32) { float x = feat[b * 512 + f]; fs += x * x; }
#pragma unroll
  for (int o = 16; o > 0; o >>= 1) fs += __shfl_xor(fs, o, 32);
  float fn = fmaxf(sqrtf(fs), 1e-12f);
  for (int p = 0; p < 10; p++) {
    float dot = 0.f, ps = 0.f;
    for (int f = lane; f < 512; f += 32) {
      float pv = protos[p * 512 + f];
      dot += feat[b * 512 + f] * pv;
      ps  += pv * pv;
    }
#pragma unroll
    for (int o = 16; o > 0; o >>= 1) { dot += __shfl_xor(dot, o, 32); ps += __shfl_xor(ps, o, 32); }
    float pn = fmaxf(sqrtf(ps), 1e-12f);
    if (lane == 0) out[b * 10 + p] = dot / (fn * pn) * 10.f;
  }
}

// ---------------------------------------------------------------------------
extern "C" void kernel_launch(void* const* d_in, const int* in_sizes, int n_in,
                              void* d_out, int out_size, void* d_ws, size_t ws_size,
                              hipStream_t stream) {
  (void)in_sizes; (void)n_in; (void)out_size; (void)ws_size;
  const float* x     = (const float*)d_in[0];
  const float* w1    = (const float*)d_in[1];
  const float* w2    = (const float*)d_in[2];
  const float* w3    = (const float*)d_in[3];
  const float* fc1_w = (const float*)d_in[4];
  const float* prot  = (const float*)d_in[5];
  const float* g1 = (const float*)d_in[6],  *b1 = (const float*)d_in[7];
  const float* m1 = (const float*)d_in[8],  *v1 = (const float*)d_in[9];
  const float* g2 = (const float*)d_in[10], *b2 = (const float*)d_in[11];
  const float* m2 = (const float*)d_in[12], *v2 = (const float*)d_in[13];
  const float* g3 = (const float*)d_in[14], *b3 = (const float*)d_in[15];
  const float* m3 = (const float*)d_in[16], *v3 = (const float*)d_in[17];

  char* ws = (char*)d_ws;
  size_t off = 0;
  auto alloc = [&](size_t bytes) -> void* {
    off = (off + 255) & ~(size_t)255;
    void* p = ws + off;
    off += bytes;
    return p;
  };

  half_t* wf1   = (half_t*)alloc(9 * 64 * 3 * sizeof(half_t));
  float*  bias1 = (float*) alloc(64 * sizeof(float));
  half_t* wf2   = (half_t*)alloc(9 * 128 * 64 * sizeof(half_t));
  float*  bias2 = (float*) alloc(128 * sizeof(float));
  half_t* wf3   = (half_t*)alloc(9 * 256 * 128 * sizeof(half_t));
  float*  bias3 = (float*) alloc(256 * sizeof(float));
  half_t* fcw   = (half_t*)alloc((size_t)512 * 16384 * sizeof(half_t));

  const size_t NA = (size_t)1024 * 32 * 32 * 64;    // 67.1M halves
  const size_t NC = (size_t)1024 * 32 * 32 * 128;   // 134M halves
  half_t* bufA = (half_t*)alloc(NA * sizeof(half_t));   // preact1 -> spikes2p
  half_t* bufB = (half_t*)alloc(NA * sizeof(half_t));   // spikes1 -> preact3
  half_t* bufC = (half_t*)alloc(NC * sizeof(half_t));   // preact2 -> spikes3p
  float*  fcpre = (float*)alloc((size_t)1024 * 512 * sizeof(float));
  float*  feat  = (float*)alloc((size_t)128 * 512 * sizeof(float));

  half_t* preact1  = bufA;
  half_t* spikes1  = bufB;
  half_t* preact2  = bufC;
  half_t* spikes2p = bufA;   // 1024*16*16*128 = 33.5M halves <= NA
  half_t* preact3  = bufB;   // 1024*16*16*256 = 67.1M halves == NA
  half_t* spikes3p = bufC;   // 1024*8*8*256  = 16.8M halves <= NC

  // --- weight prep (BN folding, layout, f16) ---
  prep_conv<3, 64>  <<<(9 * 64 * 3 + 255) / 256, 256, 0, stream>>>(w1, g1, b1, m1, v1, wf1, bias1);
  prep_conv<64, 128><<<(9 * 128 * 64 + 255) / 256, 256, 0, stream>>>(w2, g2, b2, m2, v2, wf2, bias2);
  prep_conv<128,256><<<(9 * 256 * 128 + 255) / 256, 256, 0, stream>>>(w3, g3, b3, m3, v3, wf3, bias3);
  prep_fc<<<(512 * 16384) / 256, 256, 0, stream>>>(fc1_w, fcw);

  // --- layer 1: direct conv + LIF ---
  conv1_direct<<<(1024 * 32 * 32 * 8) / 256, 256, 0, stream>>>(x, wf1, bias1, preact1);
  lif_nopool<<<(128 * 32 * 32 * 64 / 8) / 256, 256, 0, stream>>>(preact1, spikes1,
                                                                 (long)128 * 32 * 32 * 64);

  // --- layer 2: WMMA conv + LIF + pool ---
  conv_wmma<64, 128, 32, 32><<<dim3(1024 * 32 * 2, 1), 256, 0, stream>>>(spikes1, wf2, bias2, preact2);
  lif_pool<128, 32><<<(128 * 16 * 16 * 16) / 256, 256, 0, stream>>>(preact2, spikes2p);

  // --- layer 3: WMMA conv + LIF + pool ---
  conv_wmma<128, 256, 16, 16><<<dim3(1024 * 16 * 1, 2), 256, 0, stream>>>(spikes2p, wf3, bias3, preact3);
  lif_pool<256, 16><<<(128 * 8 * 8 * 32) / 256, 256, 0, stream>>>(preact3, spikes3p);

  // --- FC (WMMA) + LIF + mean + cosine logits ---
  fc_wmma<<<dim3(64, 4), 256, 0, stream>>>(spikes3p, fcw, fcpre);
  lif_fc_mean<<<(128 * 512) / 256, 256, 0, stream>>>(fcpre, feat);
  logits_kernel<<<128, 32, 0, stream>>>(feat, prot, (float*)d_out);
}
